// MambaSegDecoder_85882166051452
// MI455X (gfx1250) — matrix operand order repaired and checked by reading, not compiled, not run on hardware
//
#include <hip/hip_runtime.h>
#include <hip/hip_bf16.h>
#include <math.h>

typedef __attribute__((ext_vector_type(16))) _Float16 v16h;
typedef __attribute__((ext_vector_type(8)))  float    v8f;

#define TPB 256
#define LN_EPS 1e-5f
#define BN_SCALE 0.999995000037f   /* 1/sqrt(1+1e-5) */

__device__ __forceinline__ float siluf(float x)   { return x / (1.f + __expf(-x)); }
__device__ __forceinline__ float sigmf(float x)   { return 1.f / (1.f + __expf(-x)); }

// ---------------------------------------------------------------------------
// WMMA GEMM:  C[M,N] = A[M,K](lda) * W[N,K](ldw)^T  (+ bias[n])
// One wave per 32x32 tile (2x2 of 16x16 WMMA).  Guarantees used:
// M % 32 == 0 and lda/ldw even for every call in this network.
// Interior: A via unconditional float2 loads; B staged global->LDS with
// GLOBAL_LOAD_ASYNC_TO_LDS_B64 (ASYNCcnt), converted f32->f16 on the DS read.
// Edges/K-tail: clamped-index loads + multiplicative masks (no EXEC splits).
// ---------------------------------------------------------------------------
__device__ __forceinline__ v16h frag_a_fast(const float* __restrict__ arow,
                                            int k0, int laneHi)
{
    v16h f;
    const float2* p0 = (const float2*)(arow + k0 + laneHi * 8);
    const float2* p1 = (const float2*)(arow + k0 + 16 + laneHi * 8);
    #pragma unroll
    for (int i = 0; i < 4; ++i) {
        float2 q = p0[i];
        f[2 * i]     = (_Float16)q.x;
        f[2 * i + 1] = (_Float16)q.y;
    }
    #pragma unroll
    for (int i = 0; i < 4; ++i) {
        float2 q = p1[i];
        f[8 + 2 * i]     = (_Float16)q.x;
        f[8 + 2 * i + 1] = (_Float16)q.y;
    }
    return f;
}

__device__ __forceinline__ v16h frag_a_guard(const float* __restrict__ arow,
                                             int k0, int laneHi, int K)
{
    v16h f;
    #pragma unroll
    for (int v = 0; v < 8; ++v) {
        #pragma unroll
        for (int s = 0; s < 2; ++s) {
            int k  = k0 + ((v >= 4) ? 16 : 0) + laneHi * 8 + (v & 3) * 2 + s;
            int kc = min(k, K - 1);
            float a = arow[kc] * ((k < K) ? 1.f : 0.f);
            f[2 * v + s] = (_Float16)a;
        }
    }
    return f;
}

__device__ __forceinline__ v16h frag_b_guard(const float* __restrict__ Wt, int ldw,
                                             int n0, int kb, int N, int K)
{
    v16h f;
    const int ln = threadIdx.x & 31;
    int nn = n0 + (ln & 15);
    int nc = min(nn, N - 1);
    #pragma unroll
    for (int e = 0; e < 16; ++e) {
        int   k  = kb + (e & 7) + (ln >> 4) * 8 + ((e >= 8) ? 16 : 0);
        int   kc = min(k, K - 1);
        float km = (k < K) ? 1.f : 0.f;
        f[e] = (_Float16)(Wt[(size_t)nc * ldw + kc] * (((nn < N) ? 1.f : 0.f) * km));
    }
    return f;
}

#define WMMA(a, b, c) __builtin_amdgcn_wmma_f32_16x16x32_f16(false, (a), false, (b), (short)0, (c), false, false)

__global__ __launch_bounds__(32)
void k_gemm_wmma(const float* __restrict__ A, int lda,
                 const float* __restrict__ Wt, int ldw,
                 float* __restrict__ C, int ldc,
                 int M, int N, int K, const float* __restrict__ bias)
{
    __shared__ float Bsf[32 * 32];   // B tile staged as raw f32 by the async DMA

    const int lane   = threadIdx.x & 31;
    const int laneLo = lane & 15;
    const int laneHi = lane >> 4;
    const int m0 = blockIdx.y * 32;
    const int n0 = blockIdx.x * 32;

    const float* arow0 = A + (size_t)(m0 + laneLo) * lda;
    const float* arow1 = arow0 + (size_t)16 * lda;

    v8f acc00 = {}, acc01 = {}, acc10 = {}, acc11 = {};
    const int KF = K & ~31;

    if (n0 + 32 <= N) {
        // ---------------- interior N: async global->LDS staged B ----------------
        const float* gwrow  = Wt + (size_t)(n0 + lane) * ldw;       // this lane's W row
        unsigned     ldsrow = (unsigned)(size_t)(&Bsf[lane * 32]);  // LDS aperture: offset in addr[31:0]
        for (int k0 = 0; k0 < KF; k0 += 32) {
            if (k0 + 32 < KF) {
                __builtin_prefetch(arow0 + k0 + 32, 0, 1);   // global_prefetch_b8
                __builtin_prefetch(arow1 + k0 + 32, 0, 1);
            }
            // each lane DMA-copies its 128B row slice (16 x b64, 8B-aligned)
            {
                const float* gp = gwrow + k0;
                #pragma unroll
                for (int j = 0; j < 16; ++j) {
                    unsigned     lo = ldsrow + j * 8;
                    const float* g2 = gp + j * 2;
                    asm volatile("global_load_async_to_lds_b64 %0, %1, off"
                                 :: "v"(lo), "v"(g2) : "memory");
                }
            }
            v16h a0 = frag_a_fast(arow0, k0, laneHi);
            v16h a1 = frag_a_fast(arow1, k0, laneHi);
            asm volatile("s_wait_asynccnt 0" ::: "memory");
            v16h b0, b1;   // ds_load_b32 gathers (lanes read rows 32 floats apart)
            #pragma unroll
            for (int e = 0; e < 16; ++e) {
                int kt = (e & 7) + laneHi * 8 + ((e >= 8) ? 16 : 0);
                b0[e] = (_Float16)Bsf[laneLo * 32 + kt];
                b1[e] = (_Float16)Bsf[(laneLo + 16) * 32 + kt];
            }
            acc00 = WMMA(a0, b0, acc00);
            acc01 = WMMA(a0, b1, acc01);
            acc10 = WMMA(a1, b0, acc10);
            acc11 = WMMA(a1, b1, acc11);
        }
    } else {
        // ---------------- edge N: clamped direct loads ----------------
        for (int k0 = 0; k0 < KF; k0 += 32) {
            v16h a0 = frag_a_fast(arow0, k0, laneHi);
            v16h a1 = frag_a_fast(arow1, k0, laneHi);
            int  kb = k0;
            v16h b0 = frag_b_guard(Wt, ldw, n0,      kb, N, K);
            v16h b1 = frag_b_guard(Wt, ldw, n0 + 16, kb, N, K);
            acc00 = WMMA(a0, b0, acc00);
            acc01 = WMMA(a0, b1, acc01);
            acc10 = WMMA(a1, b0, acc10);
            acc11 = WMMA(a1, b1, acc11);
        }
    }

    if (KF < K) {   // K tail (only the tiny dt projections: K = 24/12/6)
        v16h a0 = frag_a_guard(arow0, KF, laneHi, K);
        v16h a1 = frag_a_guard(arow1, KF, laneHi, K);
        int  kb = KF;
        v16h b0 = frag_b_guard(Wt, ldw, n0,      kb, N, K);
        v16h b1 = frag_b_guard(Wt, ldw, n0 + 16, kb, N, K);
        acc00 = WMMA(a0, b0, acc00);
        acc01 = WMMA(a0, b1, acc01);
        acc10 = WMMA(a1, b0, acc10);
        acc11 = WMMA(a1, b1, acc11);
    }

    // ---------------- store (M always full; guard N only) ----------------
    {
        int n = n0 + laneLo;
        if (n < N) {
            float bv = bias ? bias[n] : 0.f;
            #pragma unroll
            for (int r = 0; r < 8; ++r) {
                int m = m0 + 8 * laneHi + r;
                C[(size_t)m * ldc + n]        = acc00[r] + bv;
                C[(size_t)(m + 16) * ldc + n] = acc10[r] + bv;
            }
        }
        int n2 = n0 + 16 + laneLo;
        if (n2 < N) {
            float bv = bias ? bias[n2] : 0.f;
            #pragma unroll
            for (int r = 0; r < 8; ++r) {
                int m = m0 + 8 * laneHi + r;
                C[(size_t)m * ldc + n2]        = acc01[r] + bv;
                C[(size_t)(m + 16) * ldc + n2] = acc11[r] + bv;
            }
        }
    }
}

// ---------------------------------------------------------------------------
// Layout / elementwise kernels
// ---------------------------------------------------------------------------
__global__ void k_nchw_to_nhwc(const float* __restrict__ in, float* __restrict__ out,
                               int Bn, int C, int H, int W)
{
    size_t tot = (size_t)Bn * C * H * W;
    size_t i = (size_t)blockIdx.x * blockDim.x + threadIdx.x;
    if (i >= tot) return;
    int ch = i % C; size_t t = i / C;
    int xx = t % W; t /= W;
    int yy = t % H; int b = t / H;
    out[i] = in[(((size_t)b * C + ch) * H + yy) * W + xx];
}

__global__ void k_nhwc_to_nchw(const float* __restrict__ in, float* __restrict__ out,
                               int Bn, int C, int H, int W)
{
    size_t tot = (size_t)Bn * C * H * W;
    size_t i = (size_t)blockIdx.x * blockDim.x + threadIdx.x;
    if (i >= tot) return;
    int xx = i % W; size_t t = i / W;
    int yy = t % H; t /= H;
    int ch = t % C; int b = t / C;
    out[i] = in[(((size_t)b * H + yy) * W + xx) * C + ch];
}

__global__ void k_pixel_shuffle(const float* __restrict__ in, float* __restrict__ out,
                                int Bn, int h, int w, int p, int co)
{
    size_t tot = (size_t)Bn * h * p * w * p * co;
    size_t i = (size_t)blockIdx.x * blockDim.x + threadIdx.x;
    if (i >= tot) return;
    int cc = i % co; size_t t = i / co;
    int X = t % (w * p); t /= (w * p);
    int Y = t % (h * p); int b = t / (h * p);
    int y = Y / p, py = Y % p, xx = X / p, px = X % p;
    out[i] = in[(((size_t)b * h + y) * w + xx) * (size_t)(p * p * co)
                + (size_t)(py * p + px) * co + cc];
}

__global__ void k_layernorm(const float* __restrict__ x, float* __restrict__ y,
                            const float* __restrict__ g, const float* __restrict__ b,
                            int R, int C)
{
    int row  = blockIdx.x * (blockDim.x >> 5) + (threadIdx.x >> 5);
    int lane = threadIdx.x & 31;
    if (row >= R) return;
    const float* xr = x + (size_t)row * C;
    float s = 0.f, s2 = 0.f;
    for (int i = lane; i < C; i += 32) { float v = xr[i]; s += v; s2 += v * v; }
    for (int o = 16; o > 0; o >>= 1) { s += __shfl_xor(s, o, 32); s2 += __shfl_xor(s2, o, 32); }
    float mean = s / C;
    float var  = s2 / C - mean * mean;
    float inv  = rsqrtf(var + LN_EPS);
    float* yr = y + (size_t)row * C;
    for (int i = lane; i < C; i += 32) yr[i] = (xr[i] - mean) * inv * g[i] + b[i];
}

__global__ void k_concat2(const float* __restrict__ a, int Ca,
                          const float* __restrict__ b, int Cb,
                          float* __restrict__ out, int rows)
{
    int Ct = Ca + Cb;
    size_t tot = (size_t)rows * Ct;
    size_t i = (size_t)blockIdx.x * blockDim.x + threadIdx.x;
    if (i >= tot) return;
    int c_ = i % Ct; size_t rr = i / Ct;
    out[i] = (c_ < Ca) ? a[rr * Ca + c_] : b[rr * Cb + (c_ - Ca)];
}

__global__ void k_add(float* __restrict__ x, const float* __restrict__ t, size_t n)
{
    size_t i = (size_t)blockIdx.x * blockDim.x + threadIdx.x;
    if (i < n) x[i] += t[i];
}

__global__ void k_gate(float* __restrict__ y, const float* __restrict__ z,
                       int rows, int C, int zld, int zoff)
{
    size_t tot = (size_t)rows * C;
    size_t i = (size_t)blockIdx.x * blockDim.x + threadIdx.x;
    if (i >= tot) return;
    int c_ = i % C; size_t rr = i / C;
    y[i] *= siluf(z[rr * (size_t)zld + zoff + c_]);
}

__global__ void k_bn_relu_add(float* __restrict__ x, const float* __restrict__ t,
                              const float* __restrict__ g, const float* __restrict__ b,
                              size_t tot, int C)
{
    size_t i = (size_t)blockIdx.x * blockDim.x + threadIdx.x;
    if (i >= tot) return;
    int ch = i % C;
    float v = t[i] * (g[ch] * BN_SCALE) + b[ch];
    x[i] += (v > 0.f) ? v : 0.f;
}

__global__ void k_lsm_combine(const float* __restrict__ t3, const float* __restrict__ g3,
                              const float* __restrict__ b3,
                              const float* __restrict__ t1, const float* __restrict__ g1,
                              const float* __restrict__ b1,
                              float* __restrict__ out, size_t tot, int C)
{
    size_t i = (size_t)blockIdx.x * blockDim.x + threadIdx.x;
    if (i >= tot) return;
    int ch = i % C;
    float a = t3[i] * (g3[ch] * BN_SCALE) + b3[ch];
    float c = t1[i] * (g1[ch] * BN_SCALE) + b1[ch];
    a = fminf(fmaxf(a, 0.f), 6.f);
    c = fminf(fmaxf(c, 0.f), 6.f);
    out[i] = a + c;
}

__global__ void k_dwconv3_silu(const float* __restrict__ in, int inld,
                               const float* __restrict__ w, const float* __restrict__ bias,
                               float* __restrict__ out, int Bn, int H, int W, int C)
{
    size_t tot = (size_t)Bn * H * W * C;
    size_t i = (size_t)blockIdx.x * blockDim.x + threadIdx.x;
    if (i >= tot) return;
    int ch = i % C; size_t t = i / C;
    int xx = t % W; t /= W;
    int yy = t % H; int b = t / H;
    float s = 0.f;
    #pragma unroll
    for (int dy = -1; dy <= 1; ++dy)
        #pragma unroll
        for (int dx = -1; dx <= 1; ++dx) {
            int y2 = yy + dy, x2 = xx + dx;
            if (y2 >= 0 && y2 < H && x2 >= 0 && x2 < W)
                s += w[ch * 9 + (dy + 1) * 3 + (dx + 1)]
                   * in[(((size_t)b * H + y2) * W + x2) * (size_t)inld + ch];
        }
    s += bias[ch];
    out[i] = siluf(s);
}

__global__ void k_lsa(const float* __restrict__ x, float* __restrict__ out,
                      const float* __restrict__ w, const float* __restrict__ g,
                      const float* __restrict__ bb, int Bn, int H, int W, int C)
{
    size_t tot = (size_t)Bn * H * W * C;
    size_t i = (size_t)blockIdx.x * blockDim.x + threadIdx.x;
    if (i >= tot) return;
    int ch = i % C; size_t t = i / C;
    int xx = t % W; t /= W;
    int yy = t % H; int b = t / H;
    float s = 0.f;
    for (int dy = -3; dy <= 3; ++dy)
        for (int dx = -3; dx <= 3; ++dx) {
            int y2 = yy + dy, x2 = xx + dx;
            if (y2 >= 0 && y2 < H && x2 >= 0 && x2 < W)
                s += w[ch * 49 + (dy + 3) * 7 + (dx + 3)]
                   * x[(((size_t)b * H + y2) * W + x2) * (size_t)C + ch];
        }
    float a = sigmf(s * (g[ch] * BN_SCALE) + bb[ch]);
    out[i] = x[i] * a;
}

__global__ void k_im2col3(const float* __restrict__ x, float* __restrict__ col,
                          int Bn, int H, int W, int C)
{
    size_t tot = (size_t)Bn * H * W * 9 * C;
    size_t i = (size_t)blockIdx.x * blockDim.x + threadIdx.x;
    if (i >= tot) return;
    int e = i % (9 * C); size_t m = i / (9 * C);
    int cin = e / 9, tap = e % 9;
    int dy = tap / 3 - 1, dx = tap % 3 - 1;
    int xx = m % W; size_t t = m / W;
    int yy = t % H; int b = t / H;
    int y2 = yy + dy, x2 = xx + dx;
    float v = 0.f;
    if (y2 >= 0 && y2 < H && x2 >= 0 && x2 < W)
        v = x[(((size_t)b * H + y2) * W + x2) * (size_t)C + cin];
    col[i] = v;
}

__global__ void k_build_xs(const float* __restrict__ xc, float* __restrict__ xs,
                           int Bn, int H, int W, int di)
{
    int L = H * W;
    size_t tot = (size_t)Bn * 4 * L * di;
    size_t i = (size_t)blockIdx.x * blockDim.x + threadIdx.x;
    if (i >= tot) return;
    int d = i % di; size_t t = i / di;
    int l = t % L; t /= L;
    int k = t % 4; int b = t / 4;
    int ls = (k >= 2) ? (L - 1 - l) : l;
    int yy, xx;
    if ((k & 1) == 0) { yy = ls / W; xx = ls % W; }
    else              { xx = ls / H; yy = ls % H; }
    xs[i] = xc[(((size_t)b * H + yy) * W + xx) * (size_t)di + d];
}

// Selective scan, one wave per (b,k,d); NSL = n/32 states per lane (1/2/4),
// fully unrolled so SSM state stays register-resident; wave32 shuffle reduce.
template <int NSL>
__global__ void k_scan(const float* __restrict__ xs, const float* __restrict__ dts,
                       const float* __restrict__ xdbl,
                       const float* __restrict__ A_log, const float* __restrict__ dt_b,
                       const float* __restrict__ Dp, float* __restrict__ ys,
                       int Bn, int di, int L, int r, int cd)
{
    const int n = NSL * 32;
    int wave = blockIdx.x * (blockDim.x >> 5) + (threadIdx.x >> 5);
    int lane = threadIdx.x & 31;
    int total = Bn * 4 * di;
    if (wave >= total) return;
    int d = wave % di;
    int k = (wave / di) % 4;
    int b = wave / (4 * di);

    float h[NSL], Aa[NSL];
    #pragma unroll
    for (int j = 0; j < NSL; ++j) {
        int nn = lane + 32 * j;
        h[j]  = 0.f;
        Aa[j] = -__expf(A_log[((size_t)k * di + d) * (size_t)n + nn]);
    }
    float dtb = dt_b[k * di + d];
    float Dpv = Dp[k * di + d];

    const float* dts_bk = dts  + ((size_t)(b * 4 + k) * L) * (size_t)di;
    const float* xs_bk  = xs   + ((size_t)(b * 4 + k) * L) * (size_t)di;
    const float* xd_bk  = xdbl + ((size_t)(b * 4 + k) * L) * (size_t)cd;
    float*       ys_bk  = ys   + ((size_t)(b * 4 + k) * L) * (size_t)di;

    for (int l = 0; l < L; ++l) {
        float dt = dts_bk[(size_t)l * di + d] + dtb;
        dt = (dt > 20.f) ? dt : log1pf(__expf(dt));       // softplus
        float u = xs_bk[(size_t)l * di + d];
        const float* xdl = xd_bk + (size_t)l * cd;
        float acc = 0.f;
        #pragma unroll
        for (int j = 0; j < NSL; ++j) {
            int nn = lane + 32 * j;
            float Bt = xdl[r + nn];
            float Ct = xdl[r + n + nn];
            h[j] = __expf(dt * Aa[j]) * h[j] + dt * u * Bt;
            acc += h[j] * Ct;
        }
        for (int o = 16; o > 0; o >>= 1) acc += __shfl_xor(acc, o, 32);
        if (lane == 0) ys_bk[(size_t)l * di + d] = acc + u * Dpv;
    }
}

__global__ void k_merge(const float* __restrict__ ys, float* __restrict__ y,
                        int Bn, int H, int W, int di)
{
    int L = H * W;
    size_t tot = (size_t)Bn * L * di;
    size_t i = (size_t)blockIdx.x * blockDim.x + threadIdx.x;
    if (i >= tot) return;
    int d = i % di; size_t t = i / di;
    int l = t % L; int b = t / L;
    int yy = l / W, xx = l % W;
    int lwh = xx * H + yy;
    size_t base = (size_t)b * 4 * L * di;
    float v = ys[base + ((size_t)0 * L + l)             * di + d]
            + ys[base + ((size_t)2 * L + (L - 1 - l))   * di + d]
            + ys[base + ((size_t)1 * L + lwh)           * di + d]
            + ys[base + ((size_t)3 * L + (L - 1 - lwh)) * di + d];
    y[i] = v;
}

__global__ void k_resize(const float* __restrict__ in, int Hin, int Win, int C,
                         float* __restrict__ out, int Bn)
{
    size_t tot = (size_t)Bn * C * 128 * 128;
    size_t i = (size_t)blockIdx.x * blockDim.x + threadIdx.x;
    if (i >= tot) return;
    int xx = i % 128; size_t t = i / 128;
    int yy = t % 128; t /= 128;
    int ch = t % C; int b = t / C;
    float sx = (xx + 0.5f) * Win / 128.f - 0.5f;
    float sy = (yy + 0.5f) * Hin / 128.f - 0.5f;
    int x0 = (int)floorf(sx), y0 = (int)floorf(sy);
    float fx = sx - x0, fy = sy - y0;
    int x1 = x0 + 1, y1 = y0 + 1;
    x0 = min(max(x0, 0), Win - 1); x1 = min(max(x1, 0), Win - 1);
    y0 = min(max(y0, 0), Hin - 1); y1 = min(max(y1, 0), Hin - 1);
    #define IN(Y,X) in[(((size_t)b * Hin + (Y)) * Win + (X)) * (size_t)C + ch]
    float v = (1.f - fy) * ((1.f - fx) * IN(y0, x0) + fx * IN(y0, x1))
            +        fy  * ((1.f - fx) * IN(y1, x0) + fx * IN(y1, x1));
    #undef IN
    out[i] = v;
}

// ---------------------------------------------------------------------------
// Host orchestration
// ---------------------------------------------------------------------------
struct BlockP {
    const float *A_log, *Dp, *conv_b, *conv_w, *dt_b, *dt_w, *in_proj_w,
                *ln_b, *ln_g, *out_ln_b, *out_ln_g, *out_proj_w, *x_proj_w;
};
struct StageP {
    const float *bam_bn_b, *bam_bn_g, *bam_w;
    BlockP blk[2];
    const float *cbd_b, *cbd_w, *lsa_bn_b, *lsa_bn_g, *lsa_w,
                *lsm_bn1_b, *lsm_bn1_g, *lsm_bn3_b, *lsm_bn3_g,
                *lsm_c1_w, *lsm_c3_w, *lsm_out_w, *pe_ln_b, *pe_ln_g, *pe_w;
};

static inline dim3 grd(size_t tot)   { return dim3((unsigned)((tot + TPB - 1) / TPB)); }
static inline dim3 grdw(int rows)    { return dim3((unsigned)((rows + 7) / 8)); }

static void gemm(hipStream_t st, const float* A, int lda, const float* Wt, int ldw,
                 float* C, int ldc, int M, int N, int K, const float* bias)
{
    dim3 g((unsigned)((N + 31) / 32), (unsigned)((M + 31) / 32));
    k_gemm_wmma<<<g, 32, 0, st>>>(A, lda, Wt, ldw, C, ldc, M, N, K, bias);
}

static void scan_launch(hipStream_t st, int n, const float* xs, const float* dts,
                        const float* xdbl, const float* A_log, const float* dt_b,
                        const float* Dp, float* ys, int Bn, int di, int L, int r, int cd)
{
    dim3 g((unsigned)((Bn * 4 * di + 7) / 8));
    if (n == 128)      k_scan<4><<<g, TPB, 0, st>>>(xs, dts, xdbl, A_log, dt_b, Dp, ys, Bn, di, L, r, cd);
    else if (n == 64)  k_scan<2><<<g, TPB, 0, st>>>(xs, dts, xdbl, A_log, dt_b, Dp, ys, Bn, di, L, r, cd);
    else               k_scan<1><<<g, TPB, 0, st>>>(xs, dts, xdbl, A_log, dt_b, Dp, ys, Bn, di, L, r, cd);
}

extern "C" void kernel_launch(void* const* d_in, const int* in_sizes, int n_in,
                              void* d_out, int out_size, void* d_ws, size_t ws_size,
                              hipStream_t stream)
{
    (void)in_sizes; (void)n_in; (void)out_size; (void)ws_size;
    const float* skips[4] = { (const float*)d_in[0], (const float*)d_in[1],
                              (const float*)d_in[2], (const float*)d_in[3] };
    int pi = 6;   // after skips0..3, h_img, w_img; JAX sorted-key leaf order
    auto P = [&]() { return (const float*)d_in[pi++]; };

    const float* fe_ln_b = P(); const float* fe_ln_g = P(); const float* fe_w = P();
    const float* seg_b   = P(); const float* seg_w   = P();

    StageP st[3];
    for (int s = 0; s < 3; ++s) {
        StageP& S = st[s];
        S.bam_bn_b = P(); S.bam_bn_g = P(); S.bam_w = P();
        for (int bl = 0; bl < 2; ++bl) {
            BlockP& Bp = S.blk[bl];
            Bp.A_log = P(); Bp.Dp = P(); Bp.conv_b = P(); Bp.conv_w = P();
            Bp.dt_b = P(); Bp.dt_w = P(); Bp.in_proj_w = P();
            Bp.ln_b = P(); Bp.ln_g = P(); Bp.out_ln_b = P(); Bp.out_ln_g = P();
            Bp.out_proj_w = P(); Bp.x_proj_w = P();
        }
        S.cbd_b = P(); S.cbd_w = P();
        S.lsa_bn_b = P(); S.lsa_bn_g = P(); S.lsa_w = P();
        S.lsm_bn1_b = P(); S.lsm_bn1_g = P(); S.lsm_bn3_b = P(); S.lsm_bn3_g = P();
        S.lsm_c1_w = P(); S.lsm_c3_w = P(); S.lsm_out_w = P();
        S.pe_ln_b = P(); S.pe_ln_g = P(); S.pe_w = P();
    }

    float* ws = (float*)d_ws;
    size_t off = 0;
    auto AL = [&](size_t nf) { float* p = ws + off; off += (nf + 63) & ~(size_t)63; return p; };

    const int Bn = 2;
    float* xb0 = AL((size_t)Bn * 32 * 32 * 96);
    float* xb1 = AL((size_t)Bn * 32 * 32 * 96);
    size_t base_mark = off;

    k_nchw_to_nhwc<<<grd((size_t)Bn * 768 * 16), TPB, 0, stream>>>(skips[3], xb0, Bn, 768, 4, 4);
    float* lres = xb0;

    const int Hs[3] = {8, 16, 32}, cs[3] = {384, 192, 96}, es_[3] = {768, 384, 192};
    const int ns_[3] = {128, 64, 32}, rs_[3] = {24, 12, 6};

    for (int s = 0; s < 3; ++s) {
        off = base_mark;
        const StageP& S = st[s];
        const int H = Hs[s], W = H, c = cs[s], e = es_[s];
        const int di = 2 * c, n = ns_[s], r = rs_[s], L = H * W;
        const int hi = H / 2, M0 = Bn * hi * hi, M = Bn * H * W;
        const int cd = r + 2 * n;

        // --- patch expand + LN ---
        float* t1 = AL((size_t)M0 * 2 * e);
        gemm(stream, lres, e, S.pe_w, e, t1, 2 * e, M0, 2 * e, e, nullptr);
        float* xe = AL((size_t)M * c);
        k_pixel_shuffle<<<grd((size_t)M * c), TPB, 0, stream>>>(t1, xe, Bn, hi, hi, 2, c);
        k_layernorm<<<grdw(M), TPB, 0, stream>>>(xe, xe, S.pe_ln_g, S.pe_ln_b, M, c);

        // --- concat skip + channel blend ---
        float* skipf = AL((size_t)M * c);
        k_nchw_to_nhwc<<<grd((size_t)M * c), TPB, 0, stream>>>(skips[2 - s], skipf, Bn, c, H, W);
        float* cat = AL((size_t)M * 2 * c);
        k_concat2<<<grd((size_t)M * 2 * c), TPB, 0, stream>>>(xe, c, skipf, c, cat, M);
        float* x = (lres == xb0) ? xb1 : xb0;
        gemm(stream, cat, 2 * c, S.cbd_w, 2 * c, x, c, M, c, 2 * c, S.cbd_b);

        // --- two SS2D blocks ---
        size_t blk_mark = off;
        for (int bl = 0; bl < 2; ++bl) {
            off = blk_mark;
            const BlockP& Bp = S.blk[bl];
            float* xn = AL((size_t)M * c);
            k_layernorm<<<grdw(M), TPB, 0, stream>>>(x, xn, Bp.ln_g, Bp.ln_b, M, c);
            float* xz = AL((size_t)M * 4 * c);
            gemm(stream, xn, c, Bp.in_proj_w, c, xz, 4 * c, M, 4 * c, c, nullptr);
            float* xc = AL((size_t)M * di);
            k_dwconv3_silu<<<grd((size_t)M * di), TPB, 0, stream>>>(xz, 4 * c, Bp.conv_w, Bp.conv_b, xc, Bn, H, W, di);
            float* xs = AL((size_t)Bn * 4 * L * di);
            k_build_xs<<<grd((size_t)Bn * 4 * L * di), TPB, 0, stream>>>(xc, xs, Bn, H, W, di);

            float* xdbl = AL((size_t)Bn * 4 * L * cd);
            for (int b2 = 0; b2 < Bn; ++b2)
                for (int k = 0; k < 4; ++k)
                    gemm(stream, xs + (size_t)(b2 * 4 + k) * L * di, di,
                         Bp.x_proj_w + (size_t)k * cd * di, di,
                         xdbl + (size_t)(b2 * 4 + k) * L * cd, cd, L, cd, di, nullptr);
            float* dts = AL((size_t)Bn * 4 * L * di);
            for (int b2 = 0; b2 < Bn; ++b2)
                for (int k = 0; k < 4; ++k)
                    gemm(stream, xdbl + (size_t)(b2 * 4 + k) * L * cd, cd,
                         Bp.dt_w + (size_t)k * di * r, r,
                         dts + (size_t)(b2 * 4 + k) * L * di, di, L, di, r, nullptr);

            float* ys = AL((size_t)Bn * 4 * L * di);
            scan_launch(stream, n, xs, dts, xdbl, Bp.A_log, Bp.dt_b, Bp.Dp, ys, Bn, di, L, r, cd);

            float* ym = AL((size_t)M * di);
            k_merge<<<grd((size_t)M * di), TPB, 0, stream>>>(ys, ym, Bn, H, W, di);
            k_layernorm<<<grdw(M), TPB, 0, stream>>>(ym, ym, Bp.out_ln_g, Bp.out_ln_b, M, di);
            k_gate<<<grd((size_t)M * di), TPB, 0, stream>>>(ym, xz, M, di, 4 * c, di);
            float* tmp = AL((size_t)M * c);
            gemm(stream, ym, di, Bp.out_proj_w, di, tmp, c, M, c, di, nullptr);
            k_add<<<grd((size_t)M * c), TPB, 0, stream>>>(x, tmp, (size_t)M * c);
        }
        off = blk_mark;

        // --- LSA attention ---
        float* x2 = lres;
        k_lsa<<<grd((size_t)M * c), TPB, 0, stream>>>(x, x2, S.lsa_w, S.lsa_bn_g, S.lsa_bn_b, Bn, H, W, c);
        x = x2;

        // --- BAM dense 3x3 + residual ---
        float* col = AL((size_t)M * 9 * c);
        k_im2col3<<<grd((size_t)M * 9 * c), TPB, 0, stream>>>(x, col, Bn, H, W, c);
        float* tb = AL((size_t)M * c);
        gemm(stream, col, 9 * c, S.bam_w, 9 * c, tb, c, M, c, 9 * c, nullptr);
        k_bn_relu_add<<<grd((size_t)M * c), TPB, 0, stream>>>(x, tb, S.bam_bn_g, S.bam_bn_b, (size_t)M * c, c);

        // --- LSM head ---
        k_im2col3<<<grd((size_t)M * 9 * c), TPB, 0, stream>>>(x, col, Bn, H, W, c);
        float* t3 = AL((size_t)M * c);
        gemm(stream, col, 9 * c, S.lsm_c3_w, 9 * c, t3, c, M, c, 9 * c, nullptr);
        float* t1g = AL((size_t)M * c);
        gemm(stream, x, c, S.lsm_c1_w, c, t1g, c, M, c, c, nullptr);
        float* lsum = AL((size_t)M * c);
        k_lsm_combine<<<grd((size_t)M * c), TPB, 0, stream>>>(t3, S.lsm_bn3_g, S.lsm_bn3_b,
                                                              t1g, S.lsm_bn1_g, S.lsm_bn1_b,
                                                              lsum, (size_t)M * c, c);
        float* lo = AL((size_t)M * 6);
        gemm(stream, lsum, c, S.lsm_out_w, c, lo, 6, M, 6, c, nullptr);
        float* outp = (float*)d_out + (size_t)(s + 1) * Bn * 6 * 128 * 128;
        k_resize<<<grd((size_t)Bn * 6 * 128 * 128), TPB, 0, stream>>>(lo, H, W, 6, outp, Bn);

        lres = x;
    }

    // --- final expand + seg head ---
    off = base_mark;
    float* tf = AL((size_t)Bn * 32 * 32 * 1536);
    gemm(stream, lres, 96, fe_w, 96, tf, 1536, Bn * 32 * 32, 1536, 96, nullptr);
    float* xf = AL((size_t)Bn * 128 * 128 * 96);
    k_pixel_shuffle<<<grd((size_t)Bn * 128 * 128 * 96), TPB, 0, stream>>>(tf, xf, Bn, 32, 32, 4, 96);
    k_layernorm<<<grdw(Bn * 128 * 128), TPB, 0, stream>>>(xf, xf, fe_ln_g, fe_ln_b, Bn * 128 * 128, 96);
    float* segN = AL((size_t)Bn * 128 * 128 * 6);
    gemm(stream, xf, 96, seg_w, 96, segN, 6, Bn * 128 * 128, 6, 96, seg_b);
    k_nhwc_to_nchw<<<grd((size_t)Bn * 6 * 128 * 128), TPB, 0, stream>>>(segN, (float*)d_out, Bn, 6, 128, 128);
}
